// MultiHeadAttention_88948772700854
// MI455X (gfx1250) — compile-verified, matrix operand order checked by reading, not run on hardware
//
#include <hip/hip_runtime.h>
#include <hip/hip_bf16.h>

// ---------------------------------------------------------------------------
// MHA forward on gfx1250 (MI455X): bf16 WMMA everywhere, f32 accumulate.
//   x:[2,2048,1024] f32, Wq/Wk/Wv/Wo:[1024,1024] f32 (Linear: y = x@W^T), bo:[1024]
// Pipeline: cvt->bf16 | Q/K/V proj GEMMs | flash attention | out proj GEMM.
// GEMM: 32x64 tile / wave (8 wmma per K-step, A frags shared over B frags).
// Flash: 32 queries / wave (softmax stats fully convergent: 1 row per lane).
// ---------------------------------------------------------------------------

typedef __attribute__((ext_vector_type(4)))  __bf16 v4bf;
typedef __attribute__((ext_vector_type(8)))  __bf16 v8bf;
typedef __attribute__((ext_vector_type(16))) __bf16 v16bf;
typedef __attribute__((ext_vector_type(8)))  float  v8f;

#define KDIM  1024
#define TSEQ  2048
#define NHEAD 16
#define HDIM  64
#define NROW  4096   // b * t

// Load a 16-element bf16 fragment slice per the CDNA5 16-bit A/B VGPR layout:
// caller passes p = row_base + kt + sel*8 ; elements [0..7] = k, [8..15] = k+16.
__device__ __forceinline__ v16bf ld_frag(const __bf16* p) {
  v8bf lo = *reinterpret_cast<const v8bf*>(p);
  v8bf hi = *reinterpret_cast<const v8bf*>(p + 16);
  return __builtin_shufflevector(lo, hi, 0,1,2,3,4,5,6,7,8,9,10,11,12,13,14,15);
}

__device__ __forceinline__ v8f wmma_bf16(v16bf a, v16bf b, v8f c) {
  return __builtin_amdgcn_wmma_f32_16x16x32_bf16(
      /*neg_a=*/false, a, /*neg_b=*/false, b,
      /*c_mod=*/(short)0, c, /*reuse_a=*/false, /*reuse_b=*/false);
}

// ---------------------------------------------------------------------------
// f32 -> bf16 conversion, 4 elements / thread
// ---------------------------------------------------------------------------
__global__ void cvt_f32_bf16(const float* __restrict__ s, __bf16* __restrict__ d, int n) {
  int i = (blockIdx.x * blockDim.x + threadIdx.x) * 4;
  if (i >= n) return;
  float4 v = *reinterpret_cast<const float4*>(s + i);
  v4bf o;
  o[0] = (__bf16)v.x; o[1] = (__bf16)v.y; o[2] = (__bf16)v.z; o[3] = (__bf16)v.w;
  *reinterpret_cast<v4bf*>(d + i) = o;
}

// ---------------------------------------------------------------------------
// GEMM: C[m,n] = sum_k A[m,k] * B[n,k]   (A:[4096,1024] bf16, B:[1024,1024] bf16)
// One wave -> 32x64 tile: 2 A frags x 4 B frags = 8 WMMA per K-step.
// mode 0: store bf16 into [b,h,t,s]      (Q, K)
// mode 1: store bf16 into [b,h,s,t]      (V transposed for P@V fragments)
// mode 2: store f32 row-major + bias     (output projection)
// ---------------------------------------------------------------------------
__global__ void gemm_bf16(const __bf16* __restrict__ A, const __bf16* __restrict__ B,
                          void* __restrict__ D, const float* __restrict__ bias, int mode) {
  const int wid  = threadIdx.x >> 5;
  const int lane = threadIdx.x & 31;
  const int tile = blockIdx.x * 8 + wid;        // 2048 wave-tiles total
  const int m0 = (tile >> 4) * 32;              // 128 row tiles of 32
  const int n0 = (tile & 15) * 64;              // 16 col groups of 64
  const int r = lane & 15, sel = lane >> 4;

  v8f acc[2][4];
#pragma unroll
  for (int m = 0; m < 2; ++m)
#pragma unroll
    for (int f = 0; f < 4; ++f) acc[m][f] = (v8f){};

  const __bf16* Ar0 = A + (size_t)(m0 + r) * KDIM;
  const __bf16* Ar1 = A + (size_t)(m0 + 16 + r) * KDIM;
  for (int kt = 0; kt < KDIM; kt += 32) {
    __builtin_prefetch(Ar0 + kt + 256, 0, 1);   // global_prefetch_b8
    __builtin_prefetch(Ar1 + kt + 256, 0, 1);
    v16bf a0 = ld_frag(Ar0 + kt + sel * 8);
    v16bf a1 = ld_frag(Ar1 + kt + sel * 8);
#pragma unroll
    for (int f = 0; f < 4; ++f) {
      const __bf16* Br = B + (size_t)(n0 + f * 16 + r) * KDIM;
      __builtin_prefetch(Br + kt + 256, 0, 1);
      v16bf bf = ld_frag(Br + kt + sel * 8);
      acc[0][f] = wmma_bf16(a0, bf, acc[0][f]);
      acc[1][f] = wmma_bf16(a1, bf, acc[1][f]);
    }
  }

  // D-matrix layout: vgpr i -> row (i + sel*8), lane -> col (f*16 + r)
#pragma unroll
  for (int mt = 0; mt < 2; ++mt) {
#pragma unroll
    for (int f = 0; f < 4; ++f) {
#pragma unroll
      for (int i = 0; i < 8; ++i) {
        int m = m0 + mt * 16 + i + sel * 8;
        int n = n0 + f * 16 + r;
        float v = acc[mt][f][i];
        if (mode == 2) {
          ((float*)D)[(size_t)m * KDIM + n] = v + bias[n];
        } else {
          int b = m >> 11, t = m & (TSEQ - 1);
          int h = n >> 6, s = n & (HDIM - 1);
          size_t idx = (mode == 0)
              ? (((size_t)(b * NHEAD + h) * TSEQ + t) * HDIM + s)
              : (((size_t)(b * NHEAD + h) * HDIM + s) * TSEQ + t);
          ((__bf16*)D)[idx] = (__bf16)v;
        }
      }
    }
  }
}

// ---------------------------------------------------------------------------
// Flash attention. One wave handles a 32-row query tile of one (b,h).
// Q,K:[b,h,t,s] bf16 ; Vt:[b,h,s,t] bf16 ; Y:[b,t, h*64+s] bf16
// Block = 4 waves, each with a private LDS slice.
// Per 32-key chunk: 8 WMMA for S = QK^T, 8 WMMA for O += P@V (V frags shared
// across both query sub-tiles). Softmax stats: lane L owns query row L.
// ---------------------------------------------------------------------------
__global__ void flash_attn(const __bf16* __restrict__ Q, const __bf16* __restrict__ Kb,
                           const __bf16* __restrict__ Vt, __bf16* __restrict__ Y) {
  __shared__ float  Sbuf[4][32 * 32];
  __shared__ __bf16 Pbuf[4][32 * 32];
  __shared__ float  aBuf[4][32];

  const int wid  = threadIdx.x >> 5;
  const int lane = threadIdx.x & 31;
  const int r = lane & 15, sel = lane >> 4;
  const int gw = blockIdx.x * 4 + wid;   // 2048 waves
  const int bh = gw >> 6;                // 64 q-tiles of 32 per (b,h)
  const int t0 = (gw & 63) * 32;

  const __bf16* Qp = Q  + (size_t)bh * TSEQ * HDIM;
  const __bf16* Kp = Kb + (size_t)bh * TSEQ * HDIM;
  const __bf16* Vp = Vt + (size_t)bh * HDIM * TSEQ;

  // Q fragments: [m-subtile 0/1][s-chunk 0/1], resident for whole pass
  v16bf qa[2][2];
#pragma unroll
  for (int m = 0; m < 2; ++m) {
#pragma unroll
    for (int j = 0; j < 2; ++j)
      qa[m][j] = ld_frag(Qp + (size_t)(t0 + m * 16 + r) * HDIM + j * 32 + sel * 8);
  }

  v8f o[2][4];
#pragma unroll
  for (int m = 0; m < 2; ++m)
#pragma unroll
    for (int f = 0; f < 4; ++f) o[m][f] = (v8f){};

  float m_old = -__builtin_inff();   // lane L owns query row t0+L
  float l_old = 0.0f;
  const float sc = 0.03125f;         // 1/sqrt(K) = 1/sqrt(1024)

  float*  S  = Sbuf[wid];
  __bf16* P  = Pbuf[wid];
  float*  ab = aBuf[wid];

  for (int c = 0; c < TSEQ / 32; ++c) {
    const int u0 = c * 32;

    // S[32 q, 32 u] = Q K^T : 4 accum tiles [m][g], 8 WMMA
    v8f s[2][2];
#pragma unroll
    for (int g = 0; g < 2; ++g) {
      v16bf b0 = ld_frag(Kp + (size_t)(u0 + g * 16 + r) * HDIM +      sel * 8);
      v16bf b1 = ld_frag(Kp + (size_t)(u0 + g * 16 + r) * HDIM + 32 + sel * 8);
#pragma unroll
      for (int m = 0; m < 2; ++m) {
        s[m][g] = (v8f){};
        s[m][g] = wmma_bf16(qa[m][0], b0, s[m][g]);
        s[m][g] = wmma_bf16(qa[m][1], b1, s[m][g]);
      }
    }

    // spill scaled scores to LDS row-major [32][32]
#pragma unroll
    for (int m = 0; m < 2; ++m)
#pragma unroll
      for (int g = 0; g < 2; ++g)
#pragma unroll
        for (int i = 0; i < 8; ++i)
          S[(m * 16 + i + sel * 8) * 32 + g * 16 + r] = s[m][g][i] * sc;
    __syncthreads();

    // online softmax row statistics: lane L owns row L (fully convergent)
    {
      const float* Srow = S + lane * 32;
      float mx = m_old;
#pragma unroll 4
      for (int j = 0; j < 32; ++j) mx = fmaxf(mx, Srow[j]);
      float alpha = __expf(m_old - mx);
      float sum = 0.0f;
      __bf16* Prow = P + lane * 32;
#pragma unroll 4
      for (int j = 0; j < 32; ++j) {
        float p = __expf(Srow[j] - mx);
        sum += p;
        Prow[j] = (__bf16)p;
      }
      l_old = alpha * l_old + sum;
      m_old = mx;
      ab[lane] = alpha;
    }
    __syncthreads();

    // rescale O accumulators by per-row alpha
    float av[2][8];
#pragma unroll
    for (int m = 0; m < 2; ++m)
#pragma unroll
      for (int i = 0; i < 8; ++i) av[m][i] = ab[m * 16 + i + sel * 8];
#pragma unroll
    for (int m = 0; m < 2; ++m)
#pragma unroll
      for (int f = 0; f < 4; ++f)
#pragma unroll
        for (int i = 0; i < 8; ++i) o[m][f][i] *= av[m][i];

    // P fragments (A-layout) from LDS, then O += P @ V (V frags shared over m)
    v16bf pa0 = ld_frag(P + (0 * 16 + r) * 32 + sel * 8);
    v16bf pa1 = ld_frag(P + (1 * 16 + r) * 32 + sel * 8);
#pragma unroll
    for (int f = 0; f < 4; ++f) {
      v16bf vb = ld_frag(Vp + (size_t)(f * 16 + r) * TSEQ + u0 + sel * 8);
      o[0][f] = wmma_bf16(pa0, vb, o[0][f]);
      o[1][f] = wmma_bf16(pa1, vb, o[1][f]);
    }
  }

  // final 1/l normalization, store Y in [b,t,1024] bf16
  ab[lane] = 1.0f / l_old;
  __syncthreads();
  float lv[2][8];
#pragma unroll
  for (int m = 0; m < 2; ++m)
#pragma unroll
    for (int i = 0; i < 8; ++i) lv[m][i] = ab[m * 16 + i + sel * 8];

  const int b = bh >> 4, h = bh & 15;
#pragma unroll
  for (int m = 0; m < 2; ++m)
#pragma unroll
    for (int f = 0; f < 4; ++f)
#pragma unroll
      for (int i = 0; i < 8; ++i) {
        int row = t0 + m * 16 + i + sel * 8;
        int col = h * HDIM + f * 16 + r;
        Y[((size_t)b * TSEQ + row) * KDIM + col] = (__bf16)(o[m][f][i] * lv[m][i]);
      }
}

// ---------------------------------------------------------------------------
extern "C" void kernel_launch(void* const* d_in, const int* in_sizes, int n_in,
                              void* d_out, int out_size, void* d_ws, size_t ws_size,
                              hipStream_t stream) {
  const float* x  = (const float*)d_in[0];
  const float* Wq = (const float*)d_in[1];
  const float* Wk = (const float*)d_in[2];
  const float* Wv = (const float*)d_in[3];
  const float* Wo = (const float*)d_in[4];
  const float* bo = (const float*)d_in[5];

  char* ws = (char*)d_ws;
  const size_t SZ_X = (size_t)NROW * KDIM * sizeof(__bf16);   // 8 MB
  const size_t SZ_W = (size_t)KDIM * KDIM * sizeof(__bf16);   // 2 MB
  __bf16* xb  = (__bf16*)ws;              ws += SZ_X;
  __bf16* wqb = (__bf16*)ws;              ws += SZ_W;
  __bf16* wkb = (__bf16*)ws;              ws += SZ_W;
  __bf16* wvb = (__bf16*)ws;              ws += SZ_W;
  __bf16* wob = (__bf16*)ws;              ws += SZ_W;
  __bf16* qb  = (__bf16*)ws;              ws += SZ_X;   // [b,h,t,s]
  __bf16* kb  = (__bf16*)ws;              ws += SZ_X;   // [b,h,t,s]
  __bf16* vtb = (__bf16*)ws;              ws += SZ_X;   // [b,h,s,t]
  __bf16* yb  = (__bf16*)ws;              ws += SZ_X;   // [b,t,1024]

  const int nX = NROW * KDIM;        // 4,194,304
  const int nW = KDIM * KDIM;        // 1,048,576
  cvt_f32_bf16<<<nX / 1024, 256, 0, stream>>>(x,  xb,  nX);
  cvt_f32_bf16<<<nW / 1024, 256, 0, stream>>>(Wq, wqb, nW);
  cvt_f32_bf16<<<nW / 1024, 256, 0, stream>>>(Wk, wkb, nW);
  cvt_f32_bf16<<<nW / 1024, 256, 0, stream>>>(Wv, wvb, nW);
  cvt_f32_bf16<<<nW / 1024, 256, 0, stream>>>(Wo, wob, nW);

  // 2048 wave-tiles (128 row-tiles of 32 x 16 col-groups), 8 waves / block
  gemm_bf16<<<256, 256, 0, stream>>>(xb, wqb, qb,  nullptr, 0);
  gemm_bf16<<<256, 256, 0, stream>>>(xb, wkb, kb,  nullptr, 0);
  gemm_bf16<<<256, 256, 0, stream>>>(xb, wvb, vtb, nullptr, 1);

  // 32 (b,h) x 64 q-tiles of 32 = 2048 waves, 4 waves / block
  flash_attn<<<512, 128, 0, stream>>>(qb, kb, vtb, yb);

  gemm_bf16<<<256, 256, 0, stream>>>(yb, wob, d_out, bo, 2);
}
